// BranchNet_22471268893233
// MI455X (gfx1250) — matrix operand order repaired and checked by reading, not compiled, not run on hardware
//
#include <hip/hip_runtime.h>
#include <hip/hip_bf16.h>

#define B_CONST   4096
#define HTOT      590
#define VOCAB     8192
#define EDIM      32
#define FDIM      32
#define CWID      7
#define TPOOL     12
#define NF1       1920
#define NH        2048
#define EPSF      1e-5f

typedef _Float16 v16h __attribute__((ext_vector_type(16)));
typedef _Float16 v8h  __attribute__((ext_vector_type(8)));
typedef float    v8f  __attribute__((ext_vector_type(8)));

static __device__ __forceinline__ v16h make_v16(v8h lo, v8h hi) {
  v16h r;
#pragma unroll
  for (int j = 0; j < 8; ++j) { r[j] = lo[j]; r[8 + j] = hi[j]; }
  return r;
}

static __device__ __forceinline__ v8f wmma_f16(v16h a, v16h b, v8f c) {
  return __builtin_amdgcn_wmma_f32_16x16x32_f16(false, a, false, b, (short)0, c,
                                                false, false);
}

// ---------------- weight preparation ----------------------------------------

__global__ void quantw_kernel(const float* __restrict__ w, _Float16* __restrict__ out,
                              int count) {
  int idx = blockIdx.x * 256 + threadIdx.x;
  if (idx < count) {
    float v = fminf(fmaxf(w[idx], -1.f), 1.f);
    out[idx] = (_Float16)rintf(v);           // WBITS=2 -> scale 1 -> {-1,0,1}
  }
}

// conv_w [5][F][E][CW] (f32) -> [5][CW][F][E] (f16)
__global__ void repack_conv_kernel(const float* __restrict__ cw,
                                   _Float16* __restrict__ out) {
  int idx = blockIdx.x * 256 + threadIdx.x;  // 5*7*32*32 = 35840
  if (idx < 5 * CWID * FDIM * EDIM) {
    int e = idx & 31;
    int t = idx >> 5;
    int f = t & 31; t >>= 5;
    int c = t % CWID;
    int i = t / CWID;
    out[idx] = (_Float16)cw[(((size_t)i * FDIM + f) * EDIM + e) * CWID + c];
  }
}

// ---------------- conv (im2col-free WMMA) ------------------------------------
// PASS 0: accumulate BN1 statistics (sum / sumsq per (slice,f)).
// PASS 1: apply BN1 + tanh + avg-pool into p[(slice,b,f,tt)].

template <int PASS>
__global__ __launch_bounds__(256)
void conv_kernel(const int* __restrict__ x, const int* __restrict__ gshift,
                 const float* __restrict__ emb, const _Float16* __restrict__ cwq,
                 const float* __restrict__ convb,
                 float* __restrict__ stat_sum, float* __restrict__ stat_sq,
                 const float* __restrict__ bn1g, const float* __restrict__ bn1b,
                 float* __restrict__ pbuf,
                 int slice, int L, int pw, int Lc, int shifting) {
  __shared__ __align__(32) _Float16 s_emb[134 * EDIM];
  __shared__ __align__(32) _Float16 s_w[CWID * FDIM * EDIM];
  __shared__ float s_pool[FDIM * 48];

  const int tid = threadIdx.x;
  const int b   = blockIdx.x;
  const int t0  = blockIdx.y * 128;

  // stage repacked conv weights for this slice
  const _Float16* wsrc = cwq + (size_t)slice * CWID * FDIM * EDIM;
  for (int i = tid; i < CWID * FDIM * EDIM; i += 256) s_w[i] = wsrc[i];

  const int gs    = gshift[b];
  const int start = HTOT - L - (shifting ? (gs % pw) : 0);

  // gather + fp32->fp16 convert 134 embedding vectors into LDS
  for (int pidx = tid; pidx < 134; pidx += 256) {
    int pos = start + t0 + pidx;
    pos = pos < 0 ? 0 : (pos > HTOT - 1 ? HTOT - 1 : pos);
    const int vtok = x[(size_t)b * HTOT + pos];
    const float* er = emb + ((size_t)slice * VOCAB + vtok) * EDIM;
    _Float16* dst = &s_emb[pidx * EDIM];
#pragma unroll
    for (int e = 0; e < EDIM; ++e) dst[e] = (_Float16)er[e];
  }
  if (PASS == 1) {
    for (int i = tid; i < FDIM * 48; i += 256) s_pool[i] = 0.0f;
  }
  __syncthreads();

  const int wid  = tid >> 5;
  const int lane = tid & 31;
  const int r    = lane & 15;
  const int half = lane >> 4;

  v8f acc0, acc1;
  {
    float b0 = convb[slice * FDIM + r];
    float b1 = convb[slice * FDIM + 16 + r];
#pragma unroll
    for (int j = 0; j < 8; ++j) { acc0[j] = b0; acc1[j] = b1; }
  }

  const int trowbase = wid * 16 + r;
#pragma unroll
  for (int c = 0; c < CWID; ++c) {
    const _Float16* ebase = &s_emb[(trowbase + c) * EDIM];
    v8h lo = *(const v8h*)(ebase + half * 16);
    v8h hi = *(const v8h*)(ebase + half * 16 + 8);
    v16h afrag = make_v16(lo, hi);
    v16h bf0 = *(const v16h*)(&s_w[(c * FDIM + r) * EDIM + half * 16]);
    v16h bf1 = *(const v16h*)(&s_w[(c * FDIM + 16 + r) * EDIM + half * 16]);
    acc0 = wmma_f16(afrag, bf0, acc0);
    acc1 = wmma_f16(afrag, bf1, acc1);
  }

  if (PASS == 0) {
#pragma unroll
    for (int tile = 0; tile < 2; ++tile) {
      v8f acc = tile ? acc1 : acc0;
      int f = tile * 16 + r;
      float s = 0.f, q = 0.f;
#pragma unroll
      for (int rr = 0; rr < 8; ++rr) {
        int tA = t0 + wid * 16 + rr + half * 8;
        float v = (tA < Lc) ? acc[rr] : 0.f;
        s += v;
        q += v * v;
      }
      s += __shfl_xor(s, 16, 32);
      q += __shfl_xor(q, 16, 32);
      if (half == 0) {
        atomicAdd(&stat_sum[slice * FDIM + f], s);
        atomicAdd(&stat_sq[slice * FDIM + f], q);
      }
    }
  } else {
    const float cntInv = 1.0f / ((float)B_CONST * (float)Lc);
    const int g0 = t0 / pw;
#pragma unroll
    for (int tile = 0; tile < 2; ++tile) {
      v8f acc = tile ? acc1 : acc0;
      int f = tile * 16 + r;
      float mean = stat_sum[slice * FDIM + f] * cntInv;
      float var  = stat_sq[slice * FDIM + f] * cntInv - mean * mean;
      float sc = bn1g[slice * FDIM + f] * rsqrtf(var + EPSF);
      float sh = bn1b[slice * FDIM + f] - mean * sc;
#pragma unroll
      for (int rr = 0; rr < 8; ++rr) {
        int tA = t0 + wid * 16 + rr + half * 8;
        if (tA < Lc) {
          float v = tanhf(acc[rr] * sc + sh);
          atomicAdd(&s_pool[f * 48 + (tA / pw - g0)], v);
        }
      }
    }
    __syncthreads();
    const float pwInv = 1.0f / (float)pw;
    for (int i = tid; i < FDIM * 48; i += 256) {
      float v = s_pool[i];
      if (v != 0.0f) {
        int f = i / 48, lg = i % 48;
        int g = g0 + lg;
        if (g < TPOOL)
          atomicAdd(&pbuf[(((size_t)slice * B_CONST + b) * FDIM + f) * TPOOL + g],
                    v * pwInv);
      }
    }
  }
}

// ---------------- pooled-BN statistics + z build ----------------------------

__global__ __launch_bounds__(256)
void bnpstats_kernel(const float* __restrict__ p, float* __restrict__ mean,
                     float* __restrict__ var) {
  __shared__ float ss[256], sq[256];
  const int idx = blockIdx.x;        // i*32+f, 160 blocks
  const int i = idx >> 5, f = idx & 31;
  float s = 0.f, q = 0.f;
  for (int j = threadIdx.x; j < B_CONST * TPOOL; j += 256) {
    int b = j / TPOOL, tt = j % TPOOL;
    float v = p[(((size_t)i * B_CONST + b) * FDIM + f) * TPOOL + tt];
    s += v;
    q += v * v;
  }
  ss[threadIdx.x] = s; sq[threadIdx.x] = q;
  __syncthreads();
  for (int st = 128; st > 0; st >>= 1) {
    if (threadIdx.x < st) {
      ss[threadIdx.x] += ss[threadIdx.x + st];
      sq[threadIdx.x] += sq[threadIdx.x + st];
    }
    __syncthreads();
  }
  if (threadIdx.x == 0) {
    float inv = 1.0f / ((float)B_CONST * (float)TPOOL);
    float m = ss[0] * inv;
    mean[idx] = m;
    var[idx]  = sq[0] * inv - m * m;
  }
}

__global__ void zbuild_kernel(const float* __restrict__ p,
                              const float* __restrict__ g,
                              const float* __restrict__ be,
                              const float* __restrict__ mean,
                              const float* __restrict__ var,
                              _Float16* __restrict__ zh) {
  size_t idx = (size_t)blockIdx.x * 256 + threadIdx.x;  // B*1920 total
  int b = (int)(idx / NF1), feat = (int)(idx % NF1);
  int i = feat / 384, rr = feat % 384;
  int f = rr / TPOOL, tt = rr % TPOOL;
  float v = p[(((size_t)i * B_CONST + b) * FDIM + f) * TPOOL + tt];
  int sf = i * FDIM + f;
  float t = g[sf] * (v - mean[sf]) * rsqrtf(var[sf] + EPSF) + be[sf];
  zh[idx] = (_Float16)t;
}

// ---------------- WMMA GEMM: C = A(f16,MxK) * W(f16,NxK)^T + bias -----------
// 64x64 per wave (16 accumulators), 8-wave block tiles 128x256.

__global__ __launch_bounds__(256)
void gemm_kernel(const _Float16* __restrict__ A, const _Float16* __restrict__ W,
                 const float* __restrict__ bias, float* __restrict__ C,
                 int M, int N, int K) {
  const int tid  = threadIdx.x;
  const int wid  = tid >> 5, lane = tid & 31;
  const int r    = lane & 15, half = lane >> 4;
  const int wr   = wid & 1, wc = wid >> 1;
  const int mBase = blockIdx.y * 128 + wr * 64;
  const int nBase = blockIdx.x * 256 + wc * 64;

  v8f acc[4][4];
#pragma unroll
  for (int ni = 0; ni < 4; ++ni) {
    float bv = bias[nBase + ni * 16 + r];
#pragma unroll
    for (int mi = 0; mi < 4; ++mi)
#pragma unroll
      for (int j = 0; j < 8; ++j) acc[mi][ni][j] = bv;
  }

  const _Float16* arow[4];
  const _Float16* wrow[4];
#pragma unroll
  for (int mi = 0; mi < 4; ++mi)
    arow[mi] = A + (size_t)(mBase + mi * 16 + r) * K;
#pragma unroll
  for (int ni = 0; ni < 4; ++ni)
    wrow[ni] = W + (size_t)(nBase + ni * 16 + r) * K;

  for (int kk = 0; kk < K; kk += 32) {
    v16h af[4], bf[4];
#pragma unroll
    for (int mi = 0; mi < 4; ++mi)
      af[mi] = make_v16(*(const v8h*)(arow[mi] + kk + half * 16),
                        *(const v8h*)(arow[mi] + kk + half * 16 + 8));
#pragma unroll
    for (int ni = 0; ni < 4; ++ni)
      bf[ni] = *(const v16h*)(wrow[ni] + kk + half * 16);
#pragma unroll
    for (int mi = 0; mi < 4; ++mi)
#pragma unroll
      for (int ni = 0; ni < 4; ++ni)
        acc[mi][ni] = wmma_f16(af[mi], bf[ni], acc[mi][ni]);
  }

#pragma unroll
  for (int mi = 0; mi < 4; ++mi)
#pragma unroll
    for (int ni = 0; ni < 4; ++ni) {
      int mrow = mBase + mi * 16 + half * 8;
      int ncol = nBase + ni * 16 + r;
#pragma unroll
      for (int rr = 0; rr < 8; ++rr)
        C[(size_t)(mrow + rr) * N + ncol] = acc[mi][ni][rr];
    }
}

// ---------------- per-column batch statistics + BN/hardtanh/quant -----------

__global__ __launch_bounds__(256)
void colstats_kernel(const float* __restrict__ Y, float* __restrict__ mean,
                     float* __restrict__ var, int M, int N) {
  __shared__ float s_s[256], s_q[256];
  const int tx = threadIdx.x & 31, ty = threadIdx.x >> 5;
  const int n = blockIdx.x * 32 + tx;
  float s = 0.f, q = 0.f;
  for (int b = ty; b < M; b += 8) {
    float v = Y[(size_t)b * N + n];
    s += v;
    q += v * v;
  }
  s_s[threadIdx.x] = s; s_q[threadIdx.x] = q;
  __syncthreads();
  if (ty == 0) {
    for (int j = 1; j < 8; ++j) { s += s_s[j * 32 + tx]; q += s_q[j * 32 + tx]; }
    float m = s / (float)M;
    mean[n] = m;
    var[n]  = q / (float)M - m * m;
  }
}

__global__ void applyq_kernel(const float* __restrict__ Y,
                              const float* __restrict__ g,
                              const float* __restrict__ be,
                              const float* __restrict__ mean,
                              const float* __restrict__ var,
                              _Float16* __restrict__ out, int N) {
  size_t idx = (size_t)blockIdx.x * 256 + threadIdx.x;  // M*N total
  int n = (int)(idx % N);
  float v = Y[idx];
  float t = g[n] * (v - mean[n]) * rsqrtf(var[n] + EPSF) + be[n];
  t = fminf(fmaxf(t, -1.f), 1.f);           // hardtanh
  t = rintf(t * 3.f) * (1.f / 3.f);          // ABITS=3 quant, ties-to-even
  out[idx] = (_Float16)t;
}

// ---------------- final 2048 -> 1 dot ---------------------------------------

__global__ __launch_bounds__(256)
void fc3_kernel(const _Float16* __restrict__ y, const _Float16* __restrict__ w,
                const float* __restrict__ b3, float* __restrict__ out) {
  __shared__ float red[256];
  const int b = blockIdx.x;
  float s = 0.f;
  for (int n = threadIdx.x; n < NH; n += 256)
    s += (float)y[(size_t)b * NH + n] * (float)w[n];
  red[threadIdx.x] = s;
  __syncthreads();
  for (int st = 128; st > 0; st >>= 1) {
    if (threadIdx.x < st) red[threadIdx.x] += red[threadIdx.x + st];
    __syncthreads();
  }
  if (threadIdx.x == 0) out[b] = red[0] + b3[0];
}

// ---------------- launch ----------------------------------------------------

extern "C" void kernel_launch(void* const* d_in, const int* in_sizes, int n_in,
                              void* d_out, int out_size, void* d_ws, size_t ws_size,
                              hipStream_t stream) {
  const int*   x     = (const int*)d_in[0];
  const int*   gs    = (const int*)d_in[1];
  const float* emb   = (const float*)d_in[2];
  const float* convw = (const float*)d_in[3];
  const float* convb = (const float*)d_in[4];
  const float* bn1g  = (const float*)d_in[5];
  const float* bn1b  = (const float*)d_in[6];
  const float* bnpg  = (const float*)d_in[7];
  const float* bnpb  = (const float*)d_in[8];
  const float* W1    = (const float*)d_in[9];
  const float* b1    = (const float*)d_in[10];
  const float* g1    = (const float*)d_in[11];
  const float* be1   = (const float*)d_in[12];
  const float* W2    = (const float*)d_in[13];
  const float* b2    = (const float*)d_in[14];
  const float* g2    = (const float*)d_in[15];
  const float* be2   = (const float*)d_in[16];
  const float* W3    = (const float*)d_in[17];
  const float* b3    = (const float*)d_in[18];
  float* out = (float*)d_out;

  char* ws = (char*)d_ws;
  size_t off = 0;
  auto take = [&](size_t bytes) -> char* {
    char* p = ws + off;
    off += (bytes + 255) & ~(size_t)255;
    return p;
  };

  _Float16* w1q   = (_Float16*)take((size_t)NH * NF1 * 2);
  _Float16* w2q   = (_Float16*)take((size_t)NH * NH * 2);
  _Float16* w3q   = (_Float16*)take((size_t)NH * 2);
  _Float16* cwq   = (_Float16*)take((size_t)5 * CWID * FDIM * EDIM * 2);
  float*    bn1s  = (float*)take(5 * FDIM * 4);
  float*    bn1q  = (float*)take(5 * FDIM * 4);
  float*    pbuf  = (float*)take((size_t)5 * B_CONST * FDIM * TPOOL * 4);
  float*    bnpm  = (float*)take(5 * FDIM * 4);
  float*    bnpv  = (float*)take(5 * FDIM * 4);
  _Float16* zh    = (_Float16*)take((size_t)B_CONST * NF1 * 2);
  float*    y1    = (float*)take((size_t)B_CONST * NH * 4);
  _Float16* y1q   = (_Float16*)take((size_t)B_CONST * NH * 2);
  float*    y2    = (float*)take((size_t)B_CONST * NH * 4);
  _Float16* y2q   = (_Float16*)take((size_t)B_CONST * NH * 2);
  float*    fcm   = (float*)take(NH * 4);
  float*    fcv   = (float*)take(NH * 4);

  // zero the accumulation buffers (every call: harness does not re-zero)
  hipMemsetAsync(bn1s, 0, 5 * FDIM * 4, stream);
  hipMemsetAsync(bn1q, 0, 5 * FDIM * 4, stream);
  hipMemsetAsync(pbuf, 0, (size_t)5 * B_CONST * FDIM * TPOOL * 4, stream);

  // weight prep
  quantw_kernel<<<(NH * NF1 + 255) / 256, 256, 0, stream>>>(W1, w1q, NH * NF1);
  quantw_kernel<<<(NH * NH + 255) / 256, 256, 0, stream>>>(W2, w2q, NH * NH);
  quantw_kernel<<<(NH + 255) / 256, 256, 0, stream>>>(W3, w3q, NH);
  repack_conv_kernel<<<(5 * CWID * FDIM * EDIM + 255) / 256, 256, 0, stream>>>(convw, cwq);

  static const int Ls[5]  = {42, 78, 150, 294, 582};
  static const int PWs[5] = {3, 6, 12, 24, 48};
  static const int SHs[5] = {0, 0, 1, 1, 1};

  // pass 1: conv + BN1 statistics
  for (int i = 0; i < 5; ++i) {
    int Lc = Ls[i] - CWID + 1;
    dim3 grid(B_CONST, (Lc + 127) / 128);
    conv_kernel<0><<<grid, 256, 0, stream>>>(x, gs, emb, cwq, convb, bn1s, bn1q,
                                             bn1g, bn1b, pbuf, i, Ls[i], PWs[i],
                                             Lc, SHs[i]);
  }
  // pass 2: conv + BN1 apply + tanh + avg-pool
  for (int i = 0; i < 5; ++i) {
    int Lc = Ls[i] - CWID + 1;
    dim3 grid(B_CONST, (Lc + 127) / 128);
    conv_kernel<1><<<grid, 256, 0, stream>>>(x, gs, emb, cwq, convb, bn1s, bn1q,
                                             bn1g, bn1b, pbuf, i, Ls[i], PWs[i],
                                             Lc, SHs[i]);
  }

  // pooled BN + feature build (f16)
  bnpstats_kernel<<<5 * FDIM, 256, 0, stream>>>(pbuf, bnpm, bnpv);
  zbuild_kernel<<<(int)(((size_t)B_CONST * NF1) / 256), 256, 0, stream>>>(
      pbuf, bnpg, bnpb, bnpm, bnpv, zh);

  // FC1
  gemm_kernel<<<dim3(NH / 256, B_CONST / 128), 256, 0, stream>>>(zh, w1q, b1, y1,
                                                                 B_CONST, NH, NF1);
  colstats_kernel<<<NH / 32, 256, 0, stream>>>(y1, fcm, fcv, B_CONST, NH);
  applyq_kernel<<<(int)(((size_t)B_CONST * NH) / 256), 256, 0, stream>>>(
      y1, g1, be1, fcm, fcv, y1q, NH);

  // FC2
  gemm_kernel<<<dim3(NH / 256, B_CONST / 128), 256, 0, stream>>>(y1q, w2q, b2, y2,
                                                                 B_CONST, NH, NH);
  colstats_kernel<<<NH / 32, 256, 0, stream>>>(y2, fcm, fcv, B_CONST, NH);
  applyq_kernel<<<(int)(((size_t)B_CONST * NH) / 256), 256, 0, stream>>>(
      y2, g2, be2, fcm, fcv, y2q, NH);

  // FC3 -> output
  fc3_kernel<<<B_CONST, 256, 0, stream>>>(y2q, w3q, b3, out);
}
